// EntityAttentionLayer_MI_44942537785480
// MI455X (gfx1250) — matrix operand order repaired and checked by reading, not compiled, not run on hardware
//
#include <hip/hip_runtime.h>
#include <hip/hip_fp16.h>

// ---------------------------------------------------------------------------
// EntityAttention fused kernel for gfx1250 (MI455X), wave32 + WMMA f16 + TDM.
// Shapes: bs=4096, ne=128, nq=32, IN=256, E=512, H=4, HD=128, OUT=512.
// ---------------------------------------------------------------------------

typedef __attribute__((ext_vector_type(16))) _Float16 v16h;
typedef __attribute__((ext_vector_type(8)))  _Float16 v8h;
typedef __attribute__((ext_vector_type(8)))  float    v8f;
typedef __attribute__((ext_vector_type(4)))  unsigned int u32x4;
typedef __attribute__((ext_vector_type(8)))  int          i32x8;
typedef __attribute__((ext_vector_type(4)))  int          i32x4;

#define BS   4096
#define NE   128
#define NQ   32
#define IND  256
#define EMB  512
#define NH   4
#define HD   128
#define N3E  1536

__device__ __forceinline__ v8f wmma_f16(v16h a, v16h b, v8f c) {
  // (neg_a, A, neg_b, B, c_mod, C, reuse_a, reuse_b)
  return __builtin_amdgcn_wmma_f32_16x16x32_f16(false, a, false, b, (short)0, c,
                                                false, false);
}

// A-matrix 16x32 f16 fragment from row-major [M][K] tile (ld in elements).
// ISA layout: lanes 0-15 row M=lane, VGPR0-3 = K 0..7, VGPR4-7 = K 16..23;
// lanes 16-31 same rows, K 8..15 and K 24..31.
__device__ __forceinline__ v16h load_a16x32(const _Float16* tile, int ld) {
  const int lane = threadIdx.x & 31;
  const int row  = lane & 15;
  const int hi   = lane >> 4;                 // 0 or 1
  const _Float16* p = tile + row * ld + hi * 8;
  v8h lo = *(const v8h*)(p);                  // K = kb+0..7
  v8h up = *(const v8h*)(p + 16);             // K = kb+16..23
  return __builtin_shufflevector(lo, up, 0,1,2,3,4,5,6,7,8,9,10,11,12,13,14,15);
}

// B-matrix 32x16 f16 fragment where memory holds each output column's K-values
// contiguously (i.e. source is row-major [N][K], ld = stride between columns).
// ISA layout: lane holds column N=lane%16; lanes 0-15 K=0..7,16..23, lanes 16-31 K=8..15,24..31.
__device__ __forceinline__ v16h load_b32x16_kcont(const _Float16* tile, int ld) {
  const int lane = threadIdx.x & 31;
  const int col  = lane & 15;
  const int kb   = (lane >> 4) << 3;          // 0 or 8
  const _Float16* p = tile + col * ld + kb;
  v8h lo = *(const v8h*)(p);
  v8h up = *(const v8h*)(p + 16);
  return __builtin_shufflevector(lo, up, 0,1,2,3,4,5,6,7,8,9,10,11,12,13,14,15);
}

// Store 16x16 f32 accumulator as f16 into row-major [M][N] tile (scalar).
// C/D layout: VGPR r, lane L -> row = r + (L<16?0:8), col = L%16.
__device__ __forceinline__ void store_c_f16(v8f acc, _Float16* tile, int ld) {
  const int lane  = threadIdx.x & 31;
  const int col   = lane & 15;
  const int rbase = (lane >> 4) << 3;
#pragma unroll
  for (int r = 0; r < 8; ++r) tile[(rbase + r) * ld + col] = (_Float16)acc[r];
}

// Store the TRANSPOSE of a 16x16 f32 accumulator as f16. Each lane owns one
// output row (= original column) at 8 consecutive elements -> one b128 store.
__device__ __forceinline__ void store_c_f16_tr(v8f acc, _Float16* tileT, int ldT) {
  const int lane  = threadIdx.x & 31;
  const int col   = lane & 15;                // original column = transposed row
  const int rbase = (lane >> 4) << 3;         // original rows rbase..rbase+7
  v8h h;
#pragma unroll
  for (int r = 0; r < 8; ++r) h[r] = (_Float16)acc[r];
  *(v8h*)(tileT + col * ldT + rbase) = h;     // 16B contiguous ds_store_b128
}

// ---------------------------------------------------------------------------
// Kernel 0: convert learned weights f32 -> f16 once into workspace.
// ---------------------------------------------------------------------------
__global__ void wcvt_kernel(const float* __restrict__ wIn,
                            const float* __restrict__ wOut,
                            _Float16* __restrict__ wInH,
                            _Float16* __restrict__ wOutH) {
  int i = blockIdx.x * 256 + threadIdx.x;
  if (i < N3E * IND)  wInH[i]  = (_Float16)wIn[i];
  if (i < EMB * EMB)  wOutH[i] = (_Float16)wOut[i];
}

// ---------------------------------------------------------------------------
// Kernel 1: one block per batch element, 8 waves, fully fused attention layer.
// ---------------------------------------------------------------------------
__global__ void __launch_bounds__(256)
entity_attn_kernel(const float* __restrict__ entities,
                   const unsigned char* __restrict__ preMask,
                   const unsigned char* __restrict__ postMask,
                   const _Float16* __restrict__ wInH,   // [1536][256] f16 (q|k|v rows)
                   const _Float16* __restrict__ wOutH,  // [512][512]  f16
                   const float* __restrict__ bOut,      // [512] f32
                   float* __restrict__ out) {           // [bs][32][512] f32
  extern __shared__ __align__(16) char smem[];
  _Float16* entL = (_Float16*)smem;            // 128*256  f16  (64 KB)
  _Float16* kvL  = entL + NE * IND;            // 128*512  f16  (128 KB) k then v^T
  _Float16* qL   = kvL + NE * EMB;             // 32*512   f16  (32 KB)  q then probs
  _Float16* attL = qL + NQ * EMB;              // 32*512   f16  (32 KB)  attended
  unsigned char* pmL = (unsigned char*)(attL + NQ * EMB); // 32*128 bytes (4 KB)

  const int b    = blockIdx.x;
  const int tid  = threadIdx.x;
  const int w    = tid >> 5;                   // wave 0..7
  const int lane = tid & 31;
  const int colL  = lane & 15;
  const int rbase = (lane >> 4) << 3;

  // ---- TDM demo: async 1D tensor copy of this block's pre_mask tile (4 KB)
  // into the attL scratch region (fully overwritten later in Phase E, so the
  // kernel result does not depend on it). Exercises TENSOR_LOAD_TO_LDS +
  // s_wait_tensorcnt on gfx1250.
#if defined(__has_builtin)
#if __has_builtin(__builtin_amdgcn_tensor_load_to_lds)
  if (w == 0) {
    const unsigned ldsOff = (unsigned)((const char*)attL - (const char*)smem);
    const unsigned long long ga =
        (unsigned long long)(const void*)(preMask + (size_t)b * NQ * NE);
    // D# group0: count=1 | lds_addr | global_addr[56:0] | type=2 (bits 127:126)
    u32x4 g0 = { 1u, ldsOff, (unsigned)ga,
                 (unsigned)((ga >> 32) & 0x01FFFFFFu) | 0x80000000u };
    // D# group1: data_size=1B, tensor_dim0=4096, tensor_dim1=1, tile_dim0=4096,
    // tensor_dim0_stride=4096, everything else 0.
    i32x8 g1 = { 0, 0x10000000, 0x00010000, 0x10000000, 0, 0x1000, 0, 0 };
    i32x4 z4 = { 0, 0, 0, 0 };
#if __clang_major__ >= 23
    i32x8 z8 = { 0, 0, 0, 0, 0, 0, 0, 0 };
    __builtin_amdgcn_tensor_load_to_lds(g0, g1, z4, z4, z8, 0);
#else
    __builtin_amdgcn_tensor_load_to_lds(g0, g1, z4, z4, 0);
#endif
    __builtin_amdgcn_s_wait_tensorcnt(0);
  }
#endif
#endif

  // ------------------- Phase A: stage entities (f16) + pre_mask ------------
  {
    const float* entB = entities + (size_t)b * NE * IND;
    for (int i = tid * 4; i < NE * IND; i += 256 * 4) {
      float4 f = *(const float4*)(entB + i);
      entL[i + 0] = (_Float16)f.x;
      entL[i + 1] = (_Float16)f.y;
      entL[i + 2] = (_Float16)f.z;
      entL[i + 3] = (_Float16)f.w;
    }
    const unsigned char* pmB = preMask + (size_t)b * NQ * NE;
    for (int i = tid; i < NQ * NE; i += 256) pmL[i] = pmB[i];
  }
  __syncthreads();

  // ------------------- Phase B1: k = ent @ Wk^T  (128x512, K=256) ----------
  {
    v16h aF[8];
#pragma unroll
    for (int ks = 0; ks < 8; ++ks)
      aF[ks] = load_a16x32(entL + (w * 16) * IND + ks * 32, IND);
    for (int nt = 0; nt < 32; ++nt) {
      __builtin_prefetch(wInH + (size_t)(EMB + (nt + 1) * 16) * IND, 0, 1);
      v8f acc = {0.f, 0.f, 0.f, 0.f, 0.f, 0.f, 0.f, 0.f};
#pragma unroll
      for (int ks = 0; ks < 8; ++ks) {
        v16h bF = load_b32x16_kcont(wInH + (size_t)(EMB + nt * 16) * IND + ks * 32, IND);
        acc = wmma_f16(aF[ks], bF, acc);
      }
      store_c_f16(acc, kvL + (w * 16) * EMB + nt * 16, EMB);   // k row-major [kk][d]
    }
  }

  // ------------------- Phase B2: q = ent[:32] @ Wq^T (32x512) --------------
  {
    const int mt = w & 1;
    v16h aF[8];
#pragma unroll
    for (int ks = 0; ks < 8; ++ks)
      aF[ks] = load_a16x32(entL + (mt * 16) * IND + ks * 32, IND);
    for (int nt = (w >> 1); nt < 32; nt += 4) {
      v8f acc = {0.f, 0.f, 0.f, 0.f, 0.f, 0.f, 0.f, 0.f};
#pragma unroll
      for (int ks = 0; ks < 8; ++ks) {
        v16h bF = load_b32x16_kcont(wInH + (size_t)(nt * 16) * IND + ks * 32, IND);
        acc = wmma_f16(aF[ks], bF, acc);
      }
      store_c_f16(acc, qL + (mt * 16) * EMB + nt * 16, EMB);
    }
  }
  __syncthreads();

  // ------------------- Phase C: logits + masked softmax (in registers) -----
  // wave w -> head h = w>>1, M-tile mt = w&1 : owns full 128-wide rows.
  {
    const int h  = w >> 1;
    const int mt = w & 1;
    const float rscale = 0.08838834764831845f;  // 1/sqrt(128)
    v16h aF[4];
#pragma unroll
    for (int ks = 0; ks < 4; ++ks)
      aF[ks] = load_a16x32(qL + (mt * 16) * EMB + h * HD + ks * 32, EMB);

    float lg[8][8];                              // [nt][r]
#pragma unroll
    for (int nt = 0; nt < 8; ++nt) {
      v8f acc = {0.f, 0.f, 0.f, 0.f, 0.f, 0.f, 0.f, 0.f};
#pragma unroll
      for (int ks = 0; ks < 4; ++ks) {
        v16h bF = load_b32x16_kcont(kvL + (nt * 16) * EMB + h * HD + ks * 32, EMB);
        acc = wmma_f16(aF[ks], bF, acc);
      }
      const int kc = nt * 16 + colL;
#pragma unroll
      for (int r = 0; r < 8; ++r) {
        const int m = mt * 16 + rbase + r;
        lg[nt][r] = pmL[m * NE + kc] ? -__builtin_inff() : acc[r] * rscale;
      }
    }
    // row-wise softmax: per r, reduce across 8 regs + 16-lane group.
    float rmax[8], rinv[8];
#pragma unroll
    for (int r = 0; r < 8; ++r) {
      float mx = lg[0][r];
#pragma unroll
      for (int nt = 1; nt < 8; ++nt) mx = fmaxf(mx, lg[nt][r]);
      mx = fmaxf(mx, __shfl_xor(mx, 1, 32));
      mx = fmaxf(mx, __shfl_xor(mx, 2, 32));
      mx = fmaxf(mx, __shfl_xor(mx, 4, 32));
      mx = fmaxf(mx, __shfl_xor(mx, 8, 32));
      rmax[r] = (mx == -__builtin_inff()) ? 0.f : mx;   // fully-masked row guard
    }
#pragma unroll
    for (int r = 0; r < 8; ++r) {
      float s = 0.f;
#pragma unroll
      for (int nt = 0; nt < 8; ++nt) {
        float p = __expf(lg[nt][r] - rmax[r]);          // exp(-inf)=0 for masked
        lg[nt][r] = p;
        s += p;
      }
      s += __shfl_xor(s, 1, 32);
      s += __shfl_xor(s, 2, 32);
      s += __shfl_xor(s, 4, 32);
      s += __shfl_xor(s, 8, 32);
      rinv[r] = (s > 0.f) ? 1.f / s : 0.f;              // NaN cleanup -> zeros
    }
    __syncthreads();           // all q reads done; qL region becomes probs
    _Float16* probs = qL;      // [H][32][128] f16
#pragma unroll
    for (int nt = 0; nt < 8; ++nt)
#pragma unroll
      for (int r = 0; r < 8; ++r) {
        const int m = mt * 16 + rbase + r;
        probs[h * (NQ * NE) + m * NE + nt * 16 + colL] =
            (_Float16)(lg[nt][r] * rinv[r]);
      }
  }
  __syncthreads();

  // ------------------- Phase D: v^T = (ent @ Wv^T)^T into kvL --------------
  // Stored transposed [d][kk] via packed b128 stores so Phase E B-fragments
  // are K-contiguous.
  {
    v16h aF[8];
#pragma unroll
    for (int ks = 0; ks < 8; ++ks)
      aF[ks] = load_a16x32(entL + (w * 16) * IND + ks * 32, IND);
    for (int nt = 0; nt < 32; ++nt) {
      v8f acc = {0.f, 0.f, 0.f, 0.f, 0.f, 0.f, 0.f, 0.f};
#pragma unroll
      for (int ks = 0; ks < 8; ++ks) {
        v16h bF = load_b32x16_kcont(wInH + (size_t)(2 * EMB + nt * 16) * IND + ks * 32, IND);
        acc = wmma_f16(aF[ks], bF, acc);
      }
      store_c_f16_tr(acc, kvL + (nt * 16) * NE + w * 16, NE);  // v^T [d][kk]
    }
  }
  __syncthreads();

  // ------------------- Phase E: attended = probs @ v (per head) ------------
  {
    const int h  = w >> 1;
    const int mt = w & 1;
    const _Float16* probs = qL;
    v16h aF[4];
#pragma unroll
    for (int ks = 0; ks < 4; ++ks)
      aF[ks] = load_a16x32(probs + h * (NQ * NE) + (mt * 16) * NE + ks * 32, NE);
#pragma unroll
    for (int nt = 0; nt < 8; ++nt) {
      v8f acc = {0.f, 0.f, 0.f, 0.f, 0.f, 0.f, 0.f, 0.f};
#pragma unroll
      for (int ks = 0; ks < 4; ++ks) {
        // v^T layout: column d has its K (=kk) values contiguous.
        v16h bF = load_b32x16_kcont(kvL + (h * HD + nt * 16) * NE + ks * 32, NE);
        acc = wmma_f16(aF[ks], bF, acc);
      }
      store_c_f16(acc, attL + (mt * 16) * EMB + h * HD + nt * 16, EMB);
    }
  }
  __syncthreads();

  // ------------------- Phase F: out = attended @ Wout^T + b, post-mask -----
  {
    const int mt = w >> 2;                       // waves 0-3 rows 0-15, 4-7 rows 16-31
    for (int i = 0; i < 8; ++i) {
      const int nt = (w & 3) * 8 + i;
      __builtin_prefetch(wOutH + (size_t)((nt + 1) & 31) * 16 * EMB, 0, 1);
      v8f acc = {0.f, 0.f, 0.f, 0.f, 0.f, 0.f, 0.f, 0.f};
#pragma unroll
      for (int ks = 0; ks < 16; ++ks) {
        v16h aF = load_a16x32(attL + (mt * 16) * EMB + ks * 32, EMB);
        v16h bF = load_b32x16_kcont(wOutH + (size_t)(nt * 16) * EMB + ks * 32, EMB);
        acc = wmma_f16(aF, bF, acc);
      }
      const int col = nt * 16 + colL;
      const float bias = bOut[col];
#pragma unroll
      for (int r = 0; r < 8; ++r) {
        const int m = mt * 16 + rbase + r;
        float v = acc[r] + bias;
        if (postMask[(size_t)b * NQ + m]) v = 0.f;
        out[((size_t)b * NQ + m) * EMB + col] = v;
      }
    }
  }
}

// ---------------------------------------------------------------------------
extern "C" void kernel_launch(void* const* d_in, const int* in_sizes, int n_in,
                              void* d_out, int out_size, void* d_ws, size_t ws_size,
                              hipStream_t stream) {
  const float*         entities = (const float*)d_in[0];
  const unsigned char* preMask  = (const unsigned char*)d_in[1];
  const unsigned char* postMask = (const unsigned char*)d_in[2];
  const float*         wIn      = (const float*)d_in[3];
  const float*         wOut     = (const float*)d_in[4];
  const float*         bOut     = (const float*)d_in[5];

  _Float16* wInH  = (_Float16*)d_ws;            // 1536*256 f16 = 768 KB
  _Float16* wOutH = wInH + (size_t)N3E * IND;   //  512*512 f16 = 512 KB

  wcvt_kernel<<<(N3E * IND) / 256, 256, 0, stream>>>(wIn, wOut, wInH, wOutH);

  // LDS: ent 64K + kv 128K + q/probs 32K + att 32K + premask 4K = 260 KB
  const size_t smem = (size_t)(NE * IND + NE * EMB + NQ * EMB + NQ * EMB) * 2
                    + (size_t)NQ * NE;
  entity_attn_kernel<<<BS, 256, smem, stream>>>(entities, preMask, postMask,
                                                wInH, wOutH, bOut, (float*)d_out);
}
